// sFSMNCell_88562225643917
// MI455X (gfx1250) — matrix operand order repaired and checked by reading, not compiled, not run on hardware
//
#include <hip/hip_runtime.h>

// ---------------------------------------------------------------------------
// sFSMN cell for MI455X (gfx1250):
//   out = x @ W1 + (Σ_{k=1..32} k·x[n-k]) @ W2 + bias
// Reformulated as a single bf16 WMMA GEMM:  out = [x | y] @ [W1 ; W2] + bias
// Workspace layout (needs ~33 MB):
//   A_cat : bf16 [16384 x 1024]   (x in cols 0..511, y in cols 512..1023)
//   Wt    : bf16 [512  x 1024]    (transposed concatenated weights)
// ---------------------------------------------------------------------------

constexpr int Bsz  = 8;
constexpr int Nseq = 2048;
constexpr int Din  = 512;
constexpr int Dout = 512;
constexpr int Kcat = 2 * Din;            // 1024
constexpr int Mrows = Bsz * Nseq;        // 16384
constexpr int CH = 128;                  // n-chunk per block in prep_A
constexpr int CHUNKS = Nseq / CH;        // 16

constexpr int BM = 128, BN = 128, BK = 32;
constexpr int PITCH = 40;                // bf16 elems per LDS row (80B, padded)

typedef __attribute__((ext_vector_type(16))) __bf16 v16bf;
typedef __attribute__((ext_vector_type(8)))  float  v8f;
typedef __attribute__((ext_vector_type(4)))  int    v4i_t;

union FragU { uint4 u[2]; v16bf v; };

// float -> bf16 (round to nearest even), packed pair into a dword
__device__ __forceinline__ unsigned short f2bf(float f) {
  unsigned u = __float_as_uint(f);
  return (unsigned short)((u + 0x7FFFu + ((u >> 16) & 1u)) >> 16);
}
__device__ __forceinline__ unsigned pack_bf16x2(float a, float b) {
  return (unsigned)f2bf(a) | ((unsigned)f2bf(b) << 16);
}

// ---------------------------------------------------------------------------
// Kernel 1: A_cat = [bf16(x) | bf16(y)], y via O(1) sliding recurrence:
//   y[n+1] = y[n] + S[n] - 32*x[n-32],  S[n+1] = S[n] + x[n+1] - x[n-31]
// grid = Bsz*CHUNKS blocks x 256 threads; thread owns one float2 column pair.
// ---------------------------------------------------------------------------
__global__ __launch_bounds__(256) void prep_A_kernel(
    const float* __restrict__ x, unsigned short* __restrict__ A) {
  const int b  = blockIdx.x / CHUNKS;
  const int n0 = (blockIdx.x % CHUNKS) * CH;
  const int d2 = threadIdx.x;  // float2 column index, 0..255

  const float2* xb =
      reinterpret_cast<const float2*>(x + (size_t)b * Nseq * Din) + d2;
  const int RS = Din / 2;  // row stride in float2

  float2 y{0.f, 0.f}, S{0.f, 0.f};
  #pragma unroll 4
  for (int k = 1; k <= 32; ++k) {
    int m = n0 - k;
    if (m >= 0) { float2 v = xb[(size_t)m * RS]; y.x += k * v.x; y.y += k * v.y; }
  }
  #pragma unroll 4
  for (int j = 0; j < 32; ++j) {
    int m = n0 - j;
    if (m >= 0) { float2 v = xb[(size_t)m * RS]; S.x += v.x; S.y += v.y; }
  }

  for (int n = n0; n < n0 + CH; ++n) {
    float2 xc = xb[(size_t)n * RS];
    unsigned short* row = A + ((size_t)b * Nseq + n) * Kcat;
    *(unsigned*)(row + 2 * d2)       = pack_bf16x2(xc.x, xc.y);
    *(unsigned*)(row + Din + 2 * d2) = pack_bf16x2(y.x, y.y);

    float2 xm32 = (n >= 32)      ? xb[(size_t)(n - 32) * RS] : float2{0.f, 0.f};
    y.x += S.x - 32.f * xm32.x;
    y.y += S.y - 32.f * xm32.y;
    float2 xp1  = (n + 1 < Nseq) ? xb[(size_t)(n + 1) * RS]  : float2{0.f, 0.f};
    float2 xm31 = (n >= 31)      ? xb[(size_t)(n - 31) * RS] : float2{0.f, 0.f};
    S.x += xp1.x - xm31.x;
    S.y += xp1.y - xm31.y;
  }
}

// ---------------------------------------------------------------------------
// Kernel 2: Wt[n][k] = bf16( k < 512 ? W1[k][n] : W2[k-512][n] )
// ---------------------------------------------------------------------------
__global__ __launch_bounds__(256) void prep_W_kernel(
    const float* __restrict__ W1, const float* __restrict__ W2,
    unsigned short* __restrict__ Wt) {
  int idx = blockIdx.x * 256 + threadIdx.x;  // n*1024 + k
  int n = idx >> 10;
  int k = idx & (Kcat - 1);
  float v = (k < Din) ? W1[(size_t)k * Dout + n]
                      : W2[(size_t)(k - Din) * Dout + n];
  Wt[idx] = f2bf(v);
}

// ---------------------------------------------------------------------------
// Kernel 3: C[16384x512] = A_cat @ Wt^T + bias, bf16 WMMA, f32 accumulate.
// 8 waves/WG; WG tile 128x128; wave tile 64(M) x 32(N); BK = 32 (one WMMA
// K-step per LDS tile). Explicitly 2x-unrolled double-buffer pipeline with
// literal buffer indices so accumulators stay pinned (no phi copies).
// ---------------------------------------------------------------------------
#if defined(__HIP_DEVICE_COMPILE__) && \
    __has_builtin(__builtin_amdgcn_global_load_async_to_lds_b128)
#define HAVE_ASYNC 1
#else
#define HAVE_ASYNC 0
#endif

#if HAVE_ASYNC
// This toolchain's builtin takes generic pointers to 16-byte int vectors.
__device__ __forceinline__ void cp_async16(const unsigned short* g,
                                           unsigned short* l) {
  __builtin_amdgcn_global_load_async_to_lds_b128((v4i_t*)g, (v4i_t*)l, 0, 0);
}
#if __has_builtin(__builtin_amdgcn_s_wait_asynccnt)
#define WAIT_ASYNC0() __builtin_amdgcn_s_wait_asynccnt(0)
#else
#define WAIT_ASYNC0() asm volatile("s_wait_asynccnt 0" ::: "memory")
#endif
#endif  // HAVE_ASYNC

__global__ __launch_bounds__(256) void gemm_wmma_kernel(
    const unsigned short* __restrict__ A, const unsigned short* __restrict__ Wt,
    const float* __restrict__ bias, float* __restrict__ C) {
  __shared__ __align__(16) unsigned short As[2][BM][PITCH];
  __shared__ __align__(16) unsigned short Bs[2][BN][PITCH];

  const int tid  = threadIdx.x;
  const int m0   = blockIdx.x * BM;
  const int n0   = blockIdx.y * BN;
  const int wave = tid >> 5;
  const int lane = tid & 31;
  const int waveM = wave & 1;    // 2 waves along M
  const int waveN = wave >> 1;   // 4 waves along N
  const int lrow  = lane & 15;
  const int lhalf = lane >> 4;   // lane group 0/1

  // Global staging: thread t loads 32B of one tile row (r = t/2, half = t&1)
  const int lr = tid >> 1;
  const int lh = tid & 1;
  const unsigned short* gA = A  + (size_t)(m0 + lr) * Kcat + lh * 16;
  const unsigned short* gB = Wt + (size_t)(n0 + lr) * Kcat + lh * 16;

  v8f acc[4][2] = {};

  // Per-tile compute: A frag = {K 0..7, K 16..23} (two 16B chunks, +32B apart)
  // per the 16-bit A layout; B frag = same K order of column n (two 16B
  // chunks, +32B apart) to match A.
  auto compute = [&](int buf) {
    FragU fa[4], fb[2];
    #pragma unroll
    for (int i = 0; i < 4; ++i) {
      const uint4* p =
          (const uint4*)&As[buf][waveM * 64 + i * 16 + lrow][lhalf * 8];
      fa[i].u[0] = p[0];
      fa[i].u[1] = p[2];
    }
    #pragma unroll
    for (int j = 0; j < 2; ++j) {
      const uint4* p =
          (const uint4*)&Bs[buf][waveN * 32 + j * 16 + lrow][lhalf * 8];
      fb[j].u[0] = p[0];
      fb[j].u[1] = p[2];
    }
    #pragma unroll
    for (int i = 0; i < 4; ++i)
      #pragma unroll
      for (int j = 0; j < 2; ++j)
        acc[i][j] = __builtin_amdgcn_wmma_f32_16x16x32_bf16(
            false, fa[i].v, false, fb[j].v, (short)0, acc[i][j], false, false);
  };

  constexpr int KT = Kcat / BK;  // 32 K-tiles (even)

#if HAVE_ASYNC
  unsigned short* lA0 = &As[0][lr][lh * 16];
  unsigned short* lA1 = &As[1][lr][lh * 16];
  unsigned short* lB0 = &Bs[0][lr][lh * 16];
  unsigned short* lB1 = &Bs[1][lr][lh * 16];

  auto stage = [&](int t, int buf) {  // call with literal buf only
    const int k = t * BK;
    unsigned short* la = buf ? lA1 : lA0;
    unsigned short* lb = buf ? lB1 : lB0;
    cp_async16(gA + k, la);
    cp_async16(gA + k + 8, la + 8);
    cp_async16(gB + k, lb);
    cp_async16(gB + k + 8, lb + 8);
  };

  // Prologue: tile 0 -> buf0.
  stage(0, 0);
  WAIT_ASYNC0();
  __syncthreads();

  // Steady state: two tiles per iteration, literal buffer indices.
  #pragma unroll 1
  for (int t = 0; t < KT - 2; t += 2) {
    stage(t + 1, 1);
    compute(0);
    WAIT_ASYNC0();
    __syncthreads();
    stage(t + 2, 0);
    compute(1);
    WAIT_ASYNC0();
    __syncthreads();
  }

  // Epilogue: tiles KT-2 (in buf0) and KT-1.
  stage(KT - 1, 1);
  compute(0);
  WAIT_ASYNC0();
  __syncthreads();
  compute(1);
#else
  // Fallback: registers are the second buffer (load -> barrier -> ds_store).
  for (int t = 0; t < KT; ++t) {
    const int k = t * BK;
    uint4 ra0 = *(const uint4*)(gA + k);
    uint4 ra1 = *(const uint4*)(gA + k + 8);
    uint4 rb0 = *(const uint4*)(gB + k);
    uint4 rb1 = *(const uint4*)(gB + k + 8);
    __syncthreads();
    *(uint4*)&As[0][lr][lh * 16]     = ra0;
    *(uint4*)&As[0][lr][lh * 16 + 8] = ra1;
    *(uint4*)&Bs[0][lr][lh * 16]     = rb0;
    *(uint4*)&Bs[0][lr][lh * 16 + 8] = rb1;
    __syncthreads();
    compute(0);
  }
#endif

  // Epilogue: C/D layout -> lane = N (per 16), vgpr e = M (lhalf*8 + e).
  #pragma unroll
  for (int i = 0; i < 4; ++i) {
    #pragma unroll
    for (int j = 0; j < 2; ++j) {
      const int n  = n0 + waveN * 32 + j * 16 + lrow;
      const int mb = m0 + waveM * 64 + i * 16 + lhalf * 8;
      const float bv = bias[n];
      #pragma unroll
      for (int e = 0; e < 8; ++e)
        C[(size_t)(mb + e) * Dout + n] = acc[i][j][e] + bv;
    }
  }
}

// ---------------------------------------------------------------------------
extern "C" void kernel_launch(void* const* d_in, const int* in_sizes, int n_in,
                              void* d_out, int out_size, void* d_ws,
                              size_t ws_size, hipStream_t stream) {
  const float* x    = (const float*)d_in[0];
  const float* W1   = (const float*)d_in[1];
  const float* W2   = (const float*)d_in[2];
  const float* bias = (const float*)d_in[3];
  float* out = (float*)d_out;

  unsigned short* A  = (unsigned short*)d_ws;              // 32 MB
  unsigned short* Wt = A + (size_t)Mrows * Kcat;           // +1 MB

  prep_A_kernel<<<Bsz * CHUNKS, 256, 0, stream>>>(x, A);
  prep_W_kernel<<<(Dout * Kcat) / 256, 256, 0, stream>>>(W1, W2, Wt);
  gemm_wmma_kernel<<<dim3(Mrows / BM, Dout / BN), 256, 0, stream>>>(A, Wt, bias,
                                                                    out);
}
